// SwinSelfAttention_6485400617326
// MI455X (gfx1250) — matrix operand order repaired and checked by reading, not compiled, not run on hardware
//
#include <hip/hip_runtime.h>

// ---------------------------------------------------------------------------
// Fused Swin window self-attention for gfx1250 (MI455X), bf16 WMMA path.
// B=16, NW=256, S=64, D=256, H=8, HD=32.  One block per (b,w) window,
// 256 threads = 8 waves (wave32); wave id == head id in the attention phase.
//
// Round-2 changes: weights pre-converted to bf16 in d_ws (tiny kernel);
// X tile staged once per block into LDS as bf16 (kills per-use cvt storm);
// LDS overlay of X-stage region with the bias table; pos_bias staged via
// gfx1250 async global->LDS DMA (global_load_async_to_lds_b32 + ASYNCcnt).
// ---------------------------------------------------------------------------

typedef __attribute__((ext_vector_type(16))) __bf16 v16bf;
typedef __attribute__((ext_vector_type(8)))  __bf16 v8bf;
typedef __attribute__((ext_vector_type(4)))  __bf16 v4bf;
typedef __attribute__((ext_vector_type(8)))  float  v8f;
typedef __attribute__((ext_vector_type(4)))  float  f4;

#define SWIN_S   64
#define SWIN_D   256
#define SWIN_H   8
#define SWIN_HD  32
#define SWIN_NW  256
#define INV_SQRT_HD 0.17677669529663687f

__device__ __forceinline__ v16bf combine16(v8bf lo, v8bf hi) {
  return __builtin_shufflevector(lo, hi, 0,1,2,3,4,5,6,7,8,9,10,11,12,13,14,15);
}

// 16-bit A-operand (16x32): lane half 'hb' holds K chunks [kb..kb+7] and
// [kb+16..kb+23] of its row, kb = 8*hb (+ k-step base folded into 'kb').
__device__ __forceinline__ v16bf load_a_bf16(const __bf16* row, int kb) {
  v8bf lo = *(const v8bf*)(row + kb);
  v8bf hi = *(const v8bf*)(row + kb + 16);
  return combine16(lo, hi);
}

// 16-bit B-operand (32x16): lane half 'hb' holds K chunks [kb..kb+7] and
// [kb+16..kb+23] of its column, kb = 8*hb.  'p' points at kb.
__device__ __forceinline__ v16bf load_b_bf16(const __bf16* p) {
  v8bf lo = *(const v8bf*)(p);
  v8bf hi = *(const v8bf*)(p + 16);
  return combine16(lo, hi);
}

// ---------------------------------------------------------------------------
// Kernel 0: one-time fp32 -> bf16 weight conversion into workspace.
// Layout: ws_bf16[mat][n][k], mat 0=Q 1=K 2=V, row-major 256x256 each.
// ---------------------------------------------------------------------------
__global__ __launch_bounds__(256)
void swin_convert_weights(const float* __restrict__ wq,
                          const float* __restrict__ wk,
                          const float* __restrict__ wv,
                          __bf16* __restrict__ dst) {
  const int i = blockIdx.x * 256 + threadIdx.x;            // 0 .. 65535
  dst[i]              = (__bf16)wq[i];
  dst[65536 + i]      = (__bf16)wk[i];
  dst[2 * 65536 + i]  = (__bf16)wv[i];
}

// ---------------------------------------------------------------------------
// Kernel 1: fused projection + windowed attention.
// ---------------------------------------------------------------------------
__global__ __launch_bounds__(256)
void swin_attn_fused(const float* __restrict__ patches,
                     const __bf16* __restrict__ wbf,       // [3][256][256] bf16
                     const float* __restrict__ bq,
                     const float* __restrict__ bk,
                     const float* __restrict__ bv,
                     const float* __restrict__ pos_bias,
                     const unsigned char* __restrict__ mask,
                     float* __restrict__ out) {
  // sQ/sK row-major [s][256] bf16; sVt transposed [d][s] bf16.
  __shared__ __align__(16) __bf16 sQ[SWIN_S * SWIN_D];
  __shared__ __align__(16) __bf16 sK[SWIN_S * SWIN_D];
  __shared__ __align__(16) __bf16 sVt[SWIN_D * SWIN_S];
  // Overlay region (32 KB): phase 1 uses it as the bf16 X tile [s][256];
  // after a barrier phase 2 uses it as the f32 bias table [q][k] (16 KB).
  __shared__ __align__(16) unsigned char sOverlay[SWIN_S * SWIN_D * 2];
  __shared__ __align__(16) float sPos[232];                // 15*15 = 225 used

  __bf16* sX    = (__bf16*)sOverlay;
  float*  sBias = (float*)sOverlay;

  const int bw   = blockIdx.x;                             // b*NW + w
  const int w    = bw & (SWIN_NW - 1);
  const int tid  = threadIdx.x;
  const int wid  = tid >> 5;
  const int lane = tid & 31;
  const int half = lane >> 4;                              // 16-lane half
  const int l16  = lane & 15;

  const float* __restrict__ xbase = patches + (size_t)bw * (SWIN_S * SWIN_D);

  // ---- stage pos_bias via gfx1250 async global->LDS DMA (ASYNCcnt) --------
  if (tid < 225) {
    unsigned ldsOff = (unsigned)(unsigned long long)(uintptr_t)&sPos[tid];
    const float* gp = pos_bias + tid;
    asm volatile("global_load_async_to_lds_b32 %0, %1, off"
                 : : "v"(ldsOff), "v"(gp) : "memory");
  }

  // ---- stage X tile to LDS as bf16 (each element converted exactly once) --
  for (int idx = tid * 4; idx < SWIN_S * SWIN_D; idx += 1024) {
    f4 x = *(const f4*)(xbase + idx);
    v4bf p;
    p[0] = (__bf16)x[0]; p[1] = (__bf16)x[1];
    p[2] = (__bf16)x[2]; p[3] = (__bf16)x[3];
    *(v4bf*)(sX + idx) = p;
  }

  asm volatile("s_wait_asynccnt 0" : : : "memory");
  __syncthreads();

  // -------- Phase 1: QKV projection (bf16 WMMA, f32 accumulate) -------------
  // 48 output n-tiles of width 16 (Q:0-15, K:16-31, V:32-47), 6 per wave.
  for (int i = 0; i < 6; ++i) {
    const int g   = wid * 6 + i;
    const int mat = g >> 4;                                // 0=Q, 1=K, 2=V
    const int nt  = g & 15;
    const __bf16* __restrict__ Wb = wbf + (size_t)mat * 65536;
    const float*  __restrict__ bb = (mat == 0) ? bq : (mat == 1) ? bk : bv;
    const int n = nt * 16 + l16;                           // lane's output col

    v8f acc[4] = {};                                       // 4 m-tiles
    const __bf16* __restrict__ wrow = Wb + (size_t)n * SWIN_D + half * 8;
#pragma unroll
    for (int ks = 0; ks < 8; ++ks) {                       // K loop, 32/step
      v16bf bmat = load_b_bf16(wrow + ks * 32);
      const int kb = ks * 32 + half * 8;
#pragma unroll
      for (int mt = 0; mt < 4; ++mt) {
        v16bf a = load_a_bf16(sX + (mt * 16 + l16) * SWIN_D, kb);
        acc[mt] = __builtin_amdgcn_wmma_f32_16x16x32_bf16(
            false, a, false, bmat, (short)0, acc[mt], false, false);
      }
    }

    const float bn = bb[n];
    if (mat < 2) {
      __bf16* dst = (mat == 0) ? sQ : sK;
      const float sc = (mat == 0) ? INV_SQRT_HD : 1.0f;    // fold 1/sqrt(HD)
#pragma unroll
      for (int mt = 0; mt < 4; ++mt)
#pragma unroll
        for (int j = 0; j < 8; ++j) {
          int m = mt * 16 + half * 8 + j;
          dst[m * SWIN_D + n] = (__bf16)((acc[mt][j] + bn) * sc);
        }
    } else {
      // V stored transposed: sVt[d][s] -> attention B-operands contiguous.
#pragma unroll
      for (int mt = 0; mt < 4; ++mt) {
        v8bf p;
#pragma unroll
        for (int j = 0; j < 8; ++j) p[j] = (__bf16)(acc[mt][j] + bn);
        *(v8bf*)(sVt + n * SWIN_S + mt * 16 + half * 8) = p;
      }
    }
  }

  __syncthreads();   // sX dead from here; rebuild region as bias table

  // ---- relative-position bias table bias[q][k] from LDS-staged pos_bias ---
  for (int t = tid; t < SWIN_S * SWIN_S; t += 256) {
    int q  = t >> 6, k = t & 63;
    int ri = (q >> 3) - (k >> 3) + 7;
    int rj = (q & 7)  - (k & 7)  + 7;
    sBias[t] = sPos[ri * 15 + rj];
  }

  __syncthreads();

  // -------- Phase 2: attention, wave 'wid' == head 'wid' --------------------
  const int h = wid;
  const __bf16* qh = sQ  + h * SWIN_HD;                    // row stride D
  const __bf16* kh = sK  + h * SWIN_HD;                    // row stride D
  const __bf16* vh = sVt + h * SWIN_HD * SWIN_S;           // [hd][s], stride S

  // Transposed logits: Lt[k,q] = sum_d K[k,d]*Q[q,d]  (A = K tile, B = Q rows)
  v16bf aK[4], bQ[4];
#pragma unroll
  for (int t = 0; t < 4; ++t) {
    aK[t] = load_a_bf16(kh + (t * 16 + l16) * SWIN_D, half * 8);
    bQ[t] = load_b_bf16(qh + (t * 16 + l16) * SWIN_D + half * 8);
  }

  v8f c[4][4];                                             // [kt][qt]
#pragma unroll
  for (int kt = 0; kt < 4; ++kt)
#pragma unroll
    for (int qt = 0; qt < 4; ++qt) {
      v8f z = {};
      c[kt][qt] = __builtin_amdgcn_wmma_f32_16x16x32_bf16(
          false, aK[kt], false, bQ[qt], (short)0, z, false, false);
    }

  // bias + mask epilogue.  Element j: k = kt*16 + 8*half + j, q = qt*16+l16.
  const unsigned char* __restrict__ mbase =
      mask + ((size_t)w * SWIN_H + h) * SWIN_S * SWIN_S;
#pragma unroll
  for (int kt = 0; kt < 4; ++kt) {
    const int k0 = kt * 16 + half * 8;
#pragma unroll
    for (int qt = 0; qt < 4; ++qt) {
      const int q = qt * 16 + l16;
      const float* bp = sBias + q * SWIN_S + k0;
      f4 b0 = *(const f4*)(bp);
      f4 b1 = *(const f4*)(bp + 4);
      unsigned long long mv = *(const unsigned long long*)(mbase + q * SWIN_S + k0);
#pragma unroll
      for (int j = 0; j < 8; ++j) {
        float bl  = (j < 4) ? b0[j] : b1[j - 4];
        float val = c[kt][qt][j] + bl;
        if ((mv >> (8 * j)) & 0xffull) val = -__builtin_inff();
        c[kt][qt][j] = val;
      }
    }
  }

  // log-softmax over k per q-column; lanes L and L+16 hold complementary
  // k-halves of the same column -> one shfl_xor(16) pair reduction.
#pragma unroll
  for (int qt = 0; qt < 4; ++qt) {
    float mx = -__builtin_inff();
#pragma unroll
    for (int kt = 0; kt < 4; ++kt)
#pragma unroll
      for (int j = 0; j < 8; ++j) mx = fmaxf(mx, c[kt][qt][j]);
    mx = fmaxf(mx, __shfl_xor(mx, 16, 32));
    float sum = 0.0f;
#pragma unroll
    for (int kt = 0; kt < 4; ++kt)
#pragma unroll
      for (int j = 0; j < 8; ++j) {
        float t = c[kt][qt][j] - mx;
        c[kt][qt][j] = t;
        sum += __expf(t);
      }
    sum += __shfl_xor(sum, 16, 32);
    float lse = __logf(sum);
#pragma unroll
    for (int kt = 0; kt < 4; ++kt)
#pragma unroll
      for (int j = 0; j < 8; ++j) c[kt][qt][j] -= lse;
  }

  // P @ V.  Per-lane layout of normalized c[][] IS the bf16 A-operand layout:
  // K-step k2 -> elements 0..7 from c[2k2][qt], 8..15 from c[2k2+1][qt].
  v16bf vB[2][2];                                          // [k2][dt]
#pragma unroll
  for (int k2 = 0; k2 < 2; ++k2)
#pragma unroll
    for (int dt = 0; dt < 2; ++dt)
      vB[k2][dt] = load_b_bf16(vh + (dt * 16 + l16) * SWIN_S + k2 * 32 + half * 8);

  v8f o[4][2] = {};
#pragma unroll
  for (int qt = 0; qt < 4; ++qt)
#pragma unroll
    for (int k2 = 0; k2 < 2; ++k2) {
      v16bf a;
#pragma unroll
      for (int j = 0; j < 8; ++j) {
        a[j]     = (__bf16)c[2 * k2][qt][j];
        a[8 + j] = (__bf16)c[2 * k2 + 1][qt][j];
      }
#pragma unroll
      for (int dt = 0; dt < 2; ++dt)
        o[qt][dt] = __builtin_amdgcn_wmma_f32_16x16x32_bf16(
            false, a, false, vB[k2][dt], (short)0, o[qt][dt], false, false);
    }

  // Store fp32 output [b,w,s,h*32+d].
  float* __restrict__ obase = out + (size_t)bw * (SWIN_S * SWIN_D) + h * SWIN_HD;
#pragma unroll
  for (int qt = 0; qt < 4; ++qt)
#pragma unroll
    for (int dt = 0; dt < 2; ++dt)
#pragma unroll
      for (int j = 0; j < 8; ++j) {
        int m = qt * 16 + half * 8 + j;
        obase[(size_t)m * SWIN_D + dt * 16 + l16] = o[qt][dt][j];
      }
}

extern "C" void kernel_launch(void* const* d_in, const int* in_sizes, int n_in,
                              void* d_out, int out_size, void* d_ws, size_t ws_size,
                              hipStream_t stream) {
  (void)in_sizes; (void)n_in; (void)out_size; (void)ws_size;
  const float* patches  = (const float*)d_in[0];
  const float* wq       = (const float*)d_in[1];
  const float* bq       = (const float*)d_in[2];
  const float* wk       = (const float*)d_in[3];
  const float* bk       = (const float*)d_in[4];
  const float* wv       = (const float*)d_in[5];
  const float* bv       = (const float*)d_in[6];
  const float* pos_bias = (const float*)d_in[7];
  const unsigned char* mask = (const unsigned char*)d_in[8];
  float* out   = (float*)d_out;
  __bf16* wbf  = (__bf16*)d_ws;          // 3*256*256*2 = 384 KB scratch

  hipLaunchKernelGGL(swin_convert_weights, dim3(256), dim3(256), 0, stream,
                     wq, wk, wv, wbf);
  hipLaunchKernelGGL(swin_attn_fused, dim3(16 * SWIN_NW), dim3(256), 0, stream,
                     patches, wbf, bq, bk, bv, pos_bias, mask, out);
}
